// VectorQuantizer_78520592106196
// MI455X (gfx1250) — matrix operand order repaired and checked by reading, not compiled, not run on hardware
//
#include <hip/hip_runtime.h>
#include <cstdint>
#include <cstddef>

// ---------------------------------------------------------------------------
// VQ-VAE vector quantizer for MI455X (gfx1250, wave32, WMMA).
// S = X @ E^T fused with row-argmin, using split-bf16 (hi+lo) WMMA:
//   x.e ~= xh.eh + xh.el + xl.eh   (f32 accumulate, ~19 mantissa bits)
// Double-buffered E-chunk staging (320 KB LDS) + 3 independent WMMA chains.
// ---------------------------------------------------------------------------

typedef __attribute__((ext_vector_type(16))) __bf16 bf16x16;
typedef __attribute__((ext_vector_type(8)))  __bf16 bf16x8;
typedef __attribute__((ext_vector_type(8)))  float  f32x8;

constexpr int VQ_N = 32768;
constexpr int VQ_D = 512;
constexpr int VQ_K = 4096;
constexpr int ROWS_PER_WG = 128;        // 8 waves x 16 rows
constexpr int GEMM_THREADS = 256;
constexpr int NCHUNK = VQ_K / 16;       // 256 chunks of 16 codes
constexpr size_t ND = (size_t)VQ_N * VQ_D;

// LDS layout (dynamic shared memory) -- exactly the 320 KB workgroup cap
constexpr size_t LDS_XH   = 0;                                      // 128 KB
constexpr size_t LDS_XL   = LDS_XH + (size_t)ROWS_PER_WG * VQ_D * 2;// 128 KB
constexpr size_t LDS_EBUF = LDS_XL + (size_t)ROWS_PER_WG * VQ_D * 2;
constexpr size_t EBUF_HALF   = (size_t)16 * VQ_D * 2;               // 16 KB (hi or lo)
constexpr size_t EBUF_STRIDE = 2 * EBUF_HALF;                       // 32 KB per buffer
constexpr size_t GEMM_LDS_BYTES = LDS_EBUF + 2 * EBUF_STRIDE;       // 327680 B

static __device__ __forceinline__ bf16x16 cat8(bf16x8 a, bf16x8 b) {
    return __builtin_shufflevector(a, b, 0,1,2,3,4,5,6,7,8,9,10,11,12,13,14,15);
}

// ---------------------------------------------------------------------------
// Kernel 0: zero the accumulators / histograms used this launch.
// ---------------------------------------------------------------------------
__global__ void vq_init(float* __restrict__ histm, float* __restrict__ hists,
                        float* __restrict__ scal) {
    int tid = threadIdx.x;
    for (int i = tid; i < VQ_K; i += 256) { histm[i] = 0.0f; hists[i] = 0.0f; }
    if (tid < 4) scal[tid] = 0.0f;
}

// ---------------------------------------------------------------------------
// Kernel 1: embedding prep: E(f32) -> Eh/El (bf16 hi/lo) + enorm[k]=sum(e^2)
// ---------------------------------------------------------------------------
__global__ void vq_prep_embedding(const float* __restrict__ E,
                                  __bf16* __restrict__ Eh, __bf16* __restrict__ El,
                                  float* __restrict__ enorm) {
    __shared__ float red[128];
    const int k = blockIdx.x;
    const int t = threadIdx.x;
    const float* row = E + (size_t)k * VQ_D;
    float ss = 0.0f;
    for (int d = t; d < VQ_D; d += 128) {
        float v = row[d];
        __bf16 h = (__bf16)v;
        Eh[(size_t)k * VQ_D + d] = h;
        El[(size_t)k * VQ_D + d] = (__bf16)(v - (float)h);
        ss += v * v;
    }
    red[t] = ss;
    __syncthreads();
    for (int s = 64; s > 0; s >>= 1) {
        if (t < s) red[t] += red[t + s];
        __syncthreads();
    }
    if (t == 0) enorm[k] = red[0];
}

// ---------------------------------------------------------------------------
// Kernel 2: fused GEMM + argmin.
//   blockIdx.x : row tile (128 rows), blockIdx.y : 0 = mean, 1 = std
//   Double-buffered E chunks: next chunk's global loads are issued before the
//   WMMA block and drained to the alternate LDS buffer after it, so the L2
//   latency hides under 48 WMMAs. Three independent accumulator chains keep
//   adjacent v_wmma instructions dependence-free.
// ---------------------------------------------------------------------------
__global__ void __launch_bounds__(GEMM_THREADS, 1)
vq_argmin_gemm(const float* __restrict__ Xmean, const float* __restrict__ Xstd,
               const __bf16* __restrict__ Eh, const __bf16* __restrict__ El,
               const float* __restrict__ enorm,
               int* __restrict__ inds_mean, int* __restrict__ inds_std,
               float* __restrict__ histm, float* __restrict__ hists) {
    extern __shared__ char smem[];
    __bf16* XhL = (__bf16*)(smem + LDS_XH);
    __bf16* XlL = (__bf16*)(smem + LDS_XL);

    const float* X   = (blockIdx.y == 0) ? Xmean : Xstd;
    int*   inds      = (blockIdx.y == 0) ? inds_mean : inds_std;
    float* hist      = (blockIdx.y == 0) ? histm : hists;

    const int tid  = threadIdx.x;
    const int lane = tid & 31;
    const int wave = tid >> 5;
    const int rowbase = blockIdx.x * ROWS_PER_WG;

    // ---- stage X tile: f32 -> bf16 hi/lo into LDS (coalesced float4 loads)
    {
        const float4* Xg4 = (const float4*)(X + (size_t)rowbase * VQ_D);
        for (int i = tid; i < (ROWS_PER_WG * VQ_D) / 4; i += GEMM_THREADS) {
            float4 v = Xg4[i];
            float vv[4] = {v.x, v.y, v.z, v.w};
            int b = i * 4;
            #pragma unroll
            for (int j = 0; j < 4; ++j) {
                __bf16 h = (__bf16)vv[j];
                XhL[b + j] = h;
                XlL[b + j] = (__bf16)(vv[j] - (float)h);
            }
        }
    }
    // ---- stage E chunk 0 into buffer 0
    {
        const uint4* ehg = (const uint4*)Eh;
        const uint4* elg = (const uint4*)El;
        uint4* dh = (uint4*)(smem + LDS_EBUF);
        uint4* dl = (uint4*)(smem + LDS_EBUF + EBUF_HALF);
        #pragma unroll
        for (int j = 0; j < 4; ++j) {
            dh[tid + 256 * j] = ehg[tid + 256 * j];
            dl[tid + 256 * j] = elg[tid + 256 * j];
        }
    }
    __syncthreads();

    float minv[8];
    int   mini[8];
    #pragma unroll
    for (int r = 0; r < 8; ++r) { minv[r] = 3.402823466e38f; mini[r] = 0; }

    const int rloc  = wave * 16 + (lane & 15);  // A row within LDS tile
    const int apart = (lane >> 4) * 16;         // A K-offset (contiguous, matches B)
    const int bpart = (lane >> 4) * 16;         // B K-offset (contiguous, matches A)
    const int col   = lane & 15;                // code column owned by this lane

    for (int c = 0; c < NCHUNK; ++c) {
        const char* cur = smem + LDS_EBUF + (size_t)(c & 1) * EBUF_STRIDE;
        const __bf16* EhB = (const __bf16*)cur;
        const __bf16* ElB = (const __bf16*)(cur + EBUF_HALF);

        // ---- issue next chunk's global loads (held in VGPRs during compute)
        uint4 teh[4], tel[4];
        const bool have_next = (c + 1 < NCHUNK);
        if (have_next) {
            const uint4* ehg = (const uint4*)(Eh + (size_t)(c + 1) * 16 * VQ_D);
            const uint4* elg = (const uint4*)(El + (size_t)(c + 1) * 16 * VQ_D);
            #pragma unroll
            for (int j = 0; j < 4; ++j) {
                teh[j] = ehg[tid + 256 * j];
                tel[j] = elg[tid + 256 * j];
            }
        }

        // ---- 16 K-steps x 3 WMMAs, three independent accumulator chains
        f32x8 acc0 = {0.f,0.f,0.f,0.f,0.f,0.f,0.f,0.f};
        f32x8 acc1 = {0.f,0.f,0.f,0.f,0.f,0.f,0.f,0.f};
        f32x8 acc2 = {0.f,0.f,0.f,0.f,0.f,0.f,0.f,0.f};
        #pragma unroll
        for (int dbase = 0; dbase < VQ_D; dbase += 32) {
            const __bf16* xh = XhL + (size_t)rloc * VQ_D + dbase + apart;
            const __bf16* xl = XlL + (size_t)rloc * VQ_D + dbase + apart;
            bf16x16 Ah = cat8(*(const bf16x8*)xh, *(const bf16x8*)(xh + 8));
            bf16x16 Al = cat8(*(const bf16x8*)xl, *(const bf16x8*)(xl + 8));
            bf16x16 Bh = *(const bf16x16*)(EhB + (size_t)col * VQ_D + dbase + bpart);
            bf16x16 Bl = *(const bf16x16*)(ElB + (size_t)col * VQ_D + dbase + bpart);
            acc0 = __builtin_amdgcn_wmma_f32_16x16x32_bf16(false, Ah, false, Bh,
                                                           (short)0, acc0, false, false);
            acc1 = __builtin_amdgcn_wmma_f32_16x16x32_bf16(false, Ah, false, Bl,
                                                           (short)0, acc1, false, false);
            acc2 = __builtin_amdgcn_wmma_f32_16x16x32_bf16(false, Al, false, Bh,
                                                           (short)0, acc2, false, false);
        }

        // ---- drain next chunk into the alternate LDS buffer
        if (have_next) {
            char* nxt = smem + LDS_EBUF + (size_t)((c + 1) & 1) * EBUF_STRIDE;
            uint4* dh = (uint4*)nxt;
            uint4* dl = (uint4*)(nxt + EBUF_HALF);
            #pragma unroll
            for (int j = 0; j < 4; ++j) {
                dh[tid + 256 * j] = teh[j];
                dl[tid + 256 * j] = tel[j];
            }
        }

        // ---- running per-lane argmin:  score = ||e||^2 - 2 x.e
        f32x8 acc = acc0 + acc1 + acc2;
        const int   code = c * 16 + col;
        const float en   = enorm[code];
        #pragma unroll
        for (int r = 0; r < 8; ++r) {
            float s = en - 2.0f * acc[r];
            if (s < minv[r]) { minv[r] = s; mini[r] = code; }
        }
        __syncthreads();  // buffer handoff (copy visible; readers of nxt done)
    }

    // ---- argmin across the 16 lanes of each half (rows 0-7 / 8-15)
    #pragma unroll
    for (int off = 1; off < 16; off <<= 1) {
        #pragma unroll
        for (int r = 0; r < 8; ++r) {
            float ov = __shfl_xor(minv[r], off, 32);
            int   oi = __shfl_xor(mini[r], off, 32);
            if (ov < minv[r] || (ov == minv[r] && oi < mini[r])) {
                minv[r] = ov; mini[r] = oi;
            }
        }
    }
    if ((lane & 15) == 0) {
        int rbase = rowbase + wave * 16 + (lane >> 4) * 8;
        #pragma unroll
        for (int r = 0; r < 8; ++r) {
            inds[rbase + r] = mini[r];
            atomicAdd(&hist[mini[r]], 1.0f);
        }
    }
}

// ---------------------------------------------------------------------------
// Kernel 3: gather quantized rows + commitment MSE + per-row KL (clipped).
//   One wave per row (lane-strided for coalescing).
// ---------------------------------------------------------------------------
__global__ void vq_gather_losses(const float* __restrict__ mean_, const float* __restrict__ std_,
                                 const float* __restrict__ E,
                                 const int* __restrict__ inds_mean, const int* __restrict__ inds_std,
                                 float* __restrict__ out_qm, float* __restrict__ out_qs,
                                 float* __restrict__ scal) {
    const int lane = threadIdx.x & 31;
    const int wave = threadIdx.x >> 5;
    const int n = blockIdx.x * 8 + wave;

    const int im = inds_mean[n];
    const int is = inds_std[n];
    const float* em = E + (size_t)im * VQ_D;
    const float* es = E + (size_t)is * VQ_D;

    float cm = 0.f, cs = 0.f, t1 = 0.f, t2 = 0.f, pp = 1.f, pq = 1.f;
    #pragma unroll
    for (int j = 0; j < 16; ++j) {
        int d = lane + j * 32;
        float mu = mean_[(size_t)n * VQ_D + d];
        float sd = std_[(size_t)n * VQ_D + d];
        float qm = em[d];
        float qs = es[d];
        out_qm[(size_t)n * VQ_D + d] = qm;   // straight-through value == qm
        out_qs[(size_t)n * VQ_D + d] = qs;
        float dm = qm - mu;                  // du = mu_prior - mu_poster
        float dsv = qs - sd;
        cm += dm * dm;
        cs += dsv * dsv;
        float sp = sd * sd;                  // sigma_poster^2
        float sq = qs * qs;                  // sigma_prior^2
        float inv = 1.0f / sq;
        t1 += sp * inv;
        t2 += dm * inv * dm;
        pp *= sp;                            // det via raw f32 product (as reference)
        pq *= sq;
    }
    #pragma unroll
    for (int off = 16; off >= 1; off >>= 1) {
        cm += __shfl_xor(cm, off, 32);
        cs += __shfl_xor(cs, off, 32);
        t1 += __shfl_xor(t1, off, 32);
        t2 += __shfl_xor(t2, off, 32);
        pp *= __shfl_xor(pp, off, 32);
        pq *= __shfl_xor(pq, off, 32);
    }
    if (lane == 0) {
        float kl = 0.5f * (t1 + t2 - (float)VQ_D
                           + logf(pq + 1e-8f) - logf(pp + 1e-8f));
        kl = fminf(fmaxf(kl, 0.0f), 10.0f);
        atomicAdd(&scal[0], cm);
        atomicAdd(&scal[1], cs);
        atomicAdd(&scal[2], kl);
    }
}

// ---------------------------------------------------------------------------
// Kernel 4: perplexities + vq_loss scalars.
// ---------------------------------------------------------------------------
__global__ void vq_finalize(const float* __restrict__ histm, const float* __restrict__ hists,
                            const float* __restrict__ scal, float* __restrict__ out_scalars) {
    __shared__ float r1[512];
    __shared__ float r2[512];
    const int t = threadIdx.x;
    float a = 0.f, b = 0.f;
    const float invN = 1.0f / (float)VQ_N;
    for (int i = t; i < VQ_K; i += 512) {
        float pm = histm[i] * invN;
        float ps = hists[i] * invN;
        a += pm * logf(pm + 1e-10f);
        b += ps * logf(ps + 1e-10f);
    }
    r1[t] = a; r2[t] = b;
    __syncthreads();
    for (int s = 256; s > 0; s >>= 1) {
        if (t < s) { r1[t] += r1[t + s]; r2[t] += r2[t + s]; }
        __syncthreads();
    }
    if (t == 0) {
        float inv_nd = 1.0f / ((float)VQ_N * (float)VQ_D);
        float commitment = scal[0] * inv_nd + scal[1] * inv_nd;
        float kl_mean = scal[2] * invN;
        out_scalars[0] = 0.25f * commitment + 1.0f * kl_mean;  // vq_loss
        out_scalars[1] = expf(-r1[0]);                         // perplexity_mean
        out_scalars[2] = expf(-r2[0]);                         // perplexity_std
    }
}

// ---------------------------------------------------------------------------
extern "C" void kernel_launch(void* const* d_in, const int* in_sizes, int n_in,
                              void* d_out, int out_size, void* d_ws, size_t ws_size,
                              hipStream_t stream) {
    (void)in_sizes; (void)n_in; (void)out_size; (void)ws_size;
    const float* latents_mean = (const float*)d_in[0];  // [N, D] f32
    const float* latents_std  = (const float*)d_in[1];  // [N, D] f32
    const float* embedding    = (const float*)d_in[2];  // [K, D] f32
    float* out = (float*)d_out;                         // qm_st | qs | 3 scalars

    // workspace layout
    char* w = (char*)d_ws;
    size_t o = 0;
    __bf16* Eh = (__bf16*)(w + o);  o += (size_t)VQ_K * VQ_D * 2;   // 4 MB
    __bf16* El = (__bf16*)(w + o);  o += (size_t)VQ_K * VQ_D * 2;   // 4 MB
    float* enorm = (float*)(w + o); o += (size_t)VQ_K * 4;          // 16 KB
    int* inds_mean = (int*)(w + o); o += (size_t)VQ_N * 4;          // 128 KB
    int* inds_std  = (int*)(w + o); o += (size_t)VQ_N * 4;          // 128 KB
    float* histm = (float*)(w + o); o += (size_t)VQ_K * 4;          // 16 KB
    float* hists = (float*)(w + o); o += (size_t)VQ_K * 4;          // 16 KB
    float* scal  = (float*)(w + o); o += 64;                        // accumulators

    // allow 320 KB dynamic LDS for the GEMM kernel (CDNA5 WGP cap)
    hipFuncSetAttribute((const void*)vq_argmin_gemm,
                        hipFuncAttributeMaxDynamicSharedMemorySize,
                        (int)GEMM_LDS_BYTES);

    vq_init<<<1, 256, 0, stream>>>(histm, hists, scal);
    vq_prep_embedding<<<VQ_K, 128, 0, stream>>>(embedding, Eh, El, enorm);

    dim3 gemm_grid(VQ_N / ROWS_PER_WG, 2);  // y: 0 = mean, 1 = std
    vq_argmin_gemm<<<gemm_grid, GEMM_THREADS, GEMM_LDS_BYTES, stream>>>(
        latents_mean, latents_std, Eh, El, enorm,
        inds_mean, inds_std, histm, hists);

    vq_gather_losses<<<VQ_N / 8, 256, 0, stream>>>(
        latents_mean, latents_std, embedding, inds_mean, inds_std,
        out, out + ND, scal);

    vq_finalize<<<1, 512, 0, stream>>>(histm, hists, scal, out + 2 * ND);
}
